// ECGGraphNetwork_2963527434791
// MI455X (gfx1250) — matrix operand order repaired and checked by reading, not compiled, not run on hardware
//
#include <hip/hip_runtime.h>

typedef __attribute__((ext_vector_type(16))) __bf16 v16bf;
typedef __attribute__((ext_vector_type(8)))  float  v8f;

#define NLEAD    12
#define BT       8                 // batch elements per block
#define ROWS     (BT * NLEAD)      // 96 rows per block
#define MT       (ROWS / 16)       // 6 M-tiles
#define FIN      512
#define HID      256
#define OUTC     128
#define NTHREADS 256

// d_ws layout (bf16, transposed N x K): W1t[256][512], W2t[256][256], W3t[128][256]
#define W1T_ELEMS (HID * FIN)
#define W2T_ELEMS (HID * HID)
#define W3T_ELEMS (OUTC * HID)
#define WST_TOTAL (W1T_ELEMS + W2T_ELEMS + W3T_ELEMS)   // 229376 elems, 458752 B

struct __align__(16) SMem {
    __bf16 H[ROWS * HID];          // 49152 B: staging for h1/h2/h3 (bf16)
    float  an[NLEAD * NLEAD];      // normalized adjacency
};

static __device__ __forceinline__ float bflo(unsigned u) { return __uint_as_float(u << 16); }
static __device__ __forceinline__ float bfhi(unsigned u) { return __uint_as_float(u & 0xffff0000u); }

// ---- A fragment (16x32 bf16, M x K) from row-major f32 global, convert to bf16
static __device__ __forceinline__ v16bf ldA_global(const float* base, int ld,
                                                   int mrow0, int k0, int lane) {
    const int m  = lane & 15;
    const int kh = (lane >> 4) << 4;                 // 0 or 16
    const float* p = base + (size_t)(mrow0 + m) * ld + k0 + kh;
    float4 f0 = ((const float4*)p)[0];
    float4 f1 = ((const float4*)p)[1];               // K = kh+0..7
    float4 f2 = ((const float4*)(p + 8))[0];
    float4 f3 = ((const float4*)(p + 8))[1];         // K = kh+8..15
    v16bf a;
    a[0]=(__bf16)f0.x;  a[1]=(__bf16)f0.y;  a[2]=(__bf16)f0.z;  a[3]=(__bf16)f0.w;
    a[4]=(__bf16)f1.x;  a[5]=(__bf16)f1.y;  a[6]=(__bf16)f1.z;  a[7]=(__bf16)f1.w;
    a[8]=(__bf16)f2.x;  a[9]=(__bf16)f2.y;  a[10]=(__bf16)f2.z; a[11]=(__bf16)f2.w;
    a[12]=(__bf16)f3.x; a[13]=(__bf16)f3.y; a[14]=(__bf16)f3.z; a[15]=(__bf16)f3.w;
    return a;
}

// ---- A fragment from row-major bf16 LDS (two 16B ds loads)
static __device__ __forceinline__ v16bf ldA_lds(const __bf16* base, int ld,
                                                int mrow0, int k0, int lane) {
    const int m  = lane & 15;
    const int kh = (lane >> 4) << 4;
    const __bf16* p = base + (mrow0 + m) * ld + k0 + kh;
    union { uint4 u[2]; v16bf v; } un;
    un.u[0] = *(const uint4*)(p);
    un.u[1] = *(const uint4*)(p + 8);
    return un.v;
}

// ---- B fragment (32x16 bf16, K x N) from PRE-TRANSPOSED bf16 (N x K) scratch:
//      per lane: column n -> row of Wt, 16 contiguous K values = two b128 loads
static __device__ __forceinline__ v16bf ldB_ws(const __bf16* wt, int ldk,
                                               int k0, int ncol0, int lane) {
    const int n  = lane & 15;
    const int kb = (lane >> 4) << 4;                 // 0 or 16
    const __bf16* p = wt + (size_t)(ncol0 + n) * ldk + k0 + kb;
    union { uint4 u[2]; v16bf v; } un;
    un.u[0] = ((const uint4*)p)[0];
    un.u[1] = ((const uint4*)p)[1];
    return un.v;
}

// ---- fallback: B fragment gathered from row-major f32 weights (no scratch)
static __device__ __forceinline__ v16bf ldB_global(const float* w, int ldn,
                                                   int k0, int ncol0, int lane) {
    const int n  = lane & 15;
    const int kb = (lane >> 4) << 4;
    const float* p = w + (size_t)(k0 + kb) * ldn + ncol0 + n;
    v16bf b;
#pragma unroll
    for (int e = 0; e < 16; ++e) b[e] = (__bf16)p[(size_t)e * ldn];
    return b;
}

// ---- store 16x16 f32 D tile as bf16 into row-major LDS
static __device__ __forceinline__ void stD_lds(__bf16* base, int ld, int mrow0,
                                               int ncol0, int lane, v8f c) {
    const int col = ncol0 + (lane & 15);
    const int mo  = (lane >> 4) << 3;                // 0 or 8
#pragma unroll
    for (int r = 0; r < 8; ++r)
        base[(mrow0 + mo + r) * ld + col] = (__bf16)c[r];
}

// ---- vectorized in-place A_norm mixing + bias (+ReLU): one (batch, 8-col) group/thread
static __device__ __forceinline__ void mix8(__bf16* Hb, const float* __restrict__ bias,
                                            bool relu, const float* an, int tid) {
    const int col0 = (tid & 31) * 8;                 // 32 groups x 8 cols = 256
    const int lb   = tid >> 5;                       // 0..7
    float v[NLEAD][8];
#pragma unroll
    for (int m = 0; m < NLEAD; ++m) {
        uint4 u = *(const uint4*)&Hb[(lb * NLEAD + m) * HID + col0];
        v[m][0] = bflo(u.x); v[m][1] = bfhi(u.x);
        v[m][2] = bflo(u.y); v[m][3] = bfhi(u.y);
        v[m][4] = bflo(u.z); v[m][5] = bfhi(u.z);
        v[m][6] = bflo(u.w); v[m][7] = bfhi(u.w);
    }
    float4 bb0 = *(const float4*)&bias[col0];
    float4 bb1 = *(const float4*)&bias[col0 + 4];
    const float bb[8] = {bb0.x, bb0.y, bb0.z, bb0.w, bb1.x, bb1.y, bb1.z, bb1.w};
#pragma unroll
    for (int l = 0; l < NLEAD; ++l) {
        float s[8];
#pragma unroll
        for (int c = 0; c < 8; ++c) s[c] = bb[c];
#pragma unroll
        for (int m = 0; m < NLEAD; ++m) {
            const float a = an[l * NLEAD + m];
#pragma unroll
            for (int c = 0; c < 8; ++c) s[c] = fmaf(a, v[m][c], s[c]);
        }
        union { __bf16 h[8]; uint4 u; } o;
#pragma unroll
        for (int c = 0; c < 8; ++c) o.h[c] = (__bf16)(relu ? fmaxf(s[c], 0.0f) : s[c]);
        *(uint4*)&Hb[(lb * NLEAD + l) * HID + col0] = o.u;
    }
}

#define WMMA_BF16(A, B, C) \
    __builtin_amdgcn_wmma_f32_16x16x32_bf16(false, (A), false, (B), (short)0, (C), false, false)

// ---- prep: convert + transpose weights f32 -> bf16 (N x K) into d_ws
__global__ __launch_bounds__(NTHREADS) void prep_weights(
    const float* __restrict__ W1, const float* __restrict__ W2,
    const float* __restrict__ W3, __bf16* __restrict__ ws)
{
    const int idx = blockIdx.x * NTHREADS + threadIdx.x;
    if (idx < W1T_ELEMS) {
        const int n = idx / FIN, k = idx % FIN;
        ws[idx] = (__bf16)W1[(size_t)k * HID + n];
    } else if (idx < W1T_ELEMS + W2T_ELEMS) {
        const int d = idx - W1T_ELEMS;
        const int n = d / HID, k = d % HID;
        ws[idx] = (__bf16)W2[(size_t)k * HID + n];
    } else if (idx < WST_TOTAL) {
        const int d = idx - W1T_ELEMS - W2T_ELEMS;
        const int n = d / HID, k = d % HID;
        ws[idx] = (__bf16)W3[(size_t)k * OUTC + n];
    }
}

template <bool PRE>
__global__ __launch_bounds__(NTHREADS) void ecg_gcn_fused(
    const float* __restrict__ x,
    const float* __restrict__ W1, const float* __restrict__ b1,
    const float* __restrict__ W2, const float* __restrict__ b2,
    const float* __restrict__ W3, const float* __restrict__ b3,
    const __bf16* __restrict__ w1t, const __bf16* __restrict__ w2t,
    const __bf16* __restrict__ w3t,
    float* __restrict__ out)
{
    __shared__ SMem sm;
    const int tid  = threadIdx.x;
    const int wave = tid >> 5;
    const int lane = tid & 31;

    // Build A_norm = D^-1/2 (adj + 2I) D^-1/2 from adjacency bitmasks (matches reference).
    if (tid < NLEAD * NLEAD) {
        const unsigned short msk[NLEAD] = {0x001E, 0x003D, 0x002B, 0x0007, 0x0003, 0x0006,
                                           0x0080, 0x0140, 0x0280, 0x0500, 0x0A00, 0x0400};
        const int i = tid / NLEAD, j = tid % NLEAD;
        const float aeff = (i == j) ? 2.0f : (float)((msk[i] >> j) & 1);
        const float di = (float)__popc((int)msk[i]) + 2.0f;
        const float dj = (float)__popc((int)msk[j]) + 2.0f;
        sm.an[tid] = aeff * rsqrtf(di * dj);
    }

    const float* xblk = x + (size_t)blockIdx.x * ROWS * FIN;
    const v8f vz = {0.f, 0.f, 0.f, 0.f, 0.f, 0.f, 0.f, 0.f};

    // ================= Layer 1: (96 x 512) @ W1 (512 x 256) =================
    {
        v8f c[MT][2];
#pragma unroll
        for (int m = 0; m < MT; ++m) { c[m][0] = vz; c[m][1] = vz; }
        const int nc0 = wave * 32;                    // 8 waves x 2 n-tiles = 256 cols
#pragma unroll 1
        for (int k0 = 0; k0 < FIN; k0 += 32) {
            v16bf bf0, bf1;
            if constexpr (PRE) {
                bf0 = ldB_ws(w1t, FIN, k0, nc0,      lane);
                bf1 = ldB_ws(w1t, FIN, k0, nc0 + 16, lane);
            } else {
                bf0 = ldB_global(W1, HID, k0, nc0,      lane);
                bf1 = ldB_global(W1, HID, k0, nc0 + 16, lane);
            }
#pragma unroll
            for (int m = 0; m < MT; ++m) {
                v16bf af = ldA_global(xblk, FIN, m * 16, k0, lane);
                c[m][0] = WMMA_BF16(af, bf0, c[m][0]);
                c[m][1] = WMMA_BF16(af, bf1, c[m][1]);
            }
        }
        __syncthreads();                              // an[] ready; H about to be written
#pragma unroll
        for (int m = 0; m < MT; ++m) {
            stD_lds(sm.H, HID, m * 16, nc0,      lane, c[m][0]);
            stD_lds(sm.H, HID, m * 16, nc0 + 16, lane, c[m][1]);
        }
    }
    __syncthreads();
    mix8(sm.H, b1, true, sm.an, tid);
    __syncthreads();

    // ================= Layer 2: (96 x 256) @ W2 (256 x 256) =================
    {
        v8f c[MT][2];
#pragma unroll
        for (int m = 0; m < MT; ++m) { c[m][0] = vz; c[m][1] = vz; }
        const int nc0 = wave * 32;
#pragma unroll 1
        for (int k0 = 0; k0 < HID; k0 += 32) {
            v16bf bf0, bf1;
            if constexpr (PRE) {
                bf0 = ldB_ws(w2t, HID, k0, nc0,      lane);
                bf1 = ldB_ws(w2t, HID, k0, nc0 + 16, lane);
            } else {
                bf0 = ldB_global(W2, HID, k0, nc0,      lane);
                bf1 = ldB_global(W2, HID, k0, nc0 + 16, lane);
            }
#pragma unroll
            for (int m = 0; m < MT; ++m) {
                v16bf af = ldA_lds(sm.H, HID, m * 16, k0, lane);
                c[m][0] = WMMA_BF16(af, bf0, c[m][0]);
                c[m][1] = WMMA_BF16(af, bf1, c[m][1]);
            }
        }
        __syncthreads();                              // all reads of H done before overwrite
#pragma unroll
        for (int m = 0; m < MT; ++m) {
            stD_lds(sm.H, HID, m * 16, nc0,      lane, c[m][0]);
            stD_lds(sm.H, HID, m * 16, nc0 + 16, lane, c[m][1]);
        }
    }
    __syncthreads();
    mix8(sm.H, b2, true, sm.an, tid);
    __syncthreads();

    // ================= Layer 3: (96 x 256) @ W3 (256 x 128) =================
    {
        v8f c[MT];
#pragma unroll
        for (int m = 0; m < MT; ++m) c[m] = vz;
        const int nc0 = wave * 16;                    // 8 waves x 1 n-tile = 128 cols
#pragma unroll 1
        for (int k0 = 0; k0 < HID; k0 += 32) {
            v16bf bf0;
            if constexpr (PRE) {
                bf0 = ldB_ws(w3t, HID, k0, nc0, lane);
            } else {
                bf0 = ldB_global(W3, OUTC, k0, nc0, lane);
            }
#pragma unroll
            for (int m = 0; m < MT; ++m) {
                v16bf af = ldA_lds(sm.H, HID, m * 16, k0, lane);
                c[m] = WMMA_BF16(af, bf0, c[m]);
            }
        }
        __syncthreads();                              // reads done before ld=128 overwrite
#pragma unroll
        for (int m = 0; m < MT; ++m)
            stD_lds(sm.H, OUTC, m * 16, nc0, lane, c[m]);
    }
    __syncthreads();

    // ===== Final: A_norm mixing + bias + mean/max pool, one (batch, 4-col) group/thread =====
    {
        const int col0 = (tid & 31) * 4;             // 32 groups x 4 cols = 128
        const int lb   = tid >> 5;                   // 0..7
        float v[NLEAD][4];
#pragma unroll
        for (int m = 0; m < NLEAD; ++m) {
            uint2 u = *(const uint2*)&sm.H[(lb * NLEAD + m) * OUTC + col0];
            v[m][0] = bflo(u.x); v[m][1] = bfhi(u.x);
            v[m][2] = bflo(u.y); v[m][3] = bfhi(u.y);
        }
        float4 bb4 = *(const float4*)&b3[col0];
        const float bb[4] = {bb4.x, bb4.y, bb4.z, bb4.w};
        float sum[4] = {0.f, 0.f, 0.f, 0.f};
        float mx[4]  = {-3.4e38f, -3.4e38f, -3.4e38f, -3.4e38f};
#pragma unroll
        for (int l = 0; l < NLEAD; ++l) {
            float s[4];
#pragma unroll
            for (int c = 0; c < 4; ++c) s[c] = bb[c];
#pragma unroll
            for (int m = 0; m < NLEAD; ++m) {
                const float a = sm.an[l * NLEAD + m];
#pragma unroll
                for (int c = 0; c < 4; ++c) s[c] = fmaf(a, v[m][c], s[c]);
            }
#pragma unroll
            for (int c = 0; c < 4; ++c) { sum[c] += s[c]; mx[c] = fmaxf(mx[c], s[c]); }
        }
        const size_t gb = (size_t)blockIdx.x * BT + lb;
        float4 mean4 = {sum[0] * (1.0f / 12.0f), sum[1] * (1.0f / 12.0f),
                        sum[2] * (1.0f / 12.0f), sum[3] * (1.0f / 12.0f)};
        float4 max4  = {mx[0], mx[1], mx[2], mx[3]};
        *(float4*)&out[gb * (2 * OUTC) + col0]        = mean4;   // mean half
        *(float4*)&out[gb * (2 * OUTC) + OUTC + col0] = max4;    // max half
    }
}

extern "C" void kernel_launch(void* const* d_in, const int* in_sizes, int n_in,
                              void* d_out, int out_size, void* d_ws, size_t ws_size,
                              hipStream_t stream) {
    const float* x  = (const float*)d_in[0];
    const float* W1 = (const float*)d_in[1];
    const float* b1 = (const float*)d_in[2];
    const float* W2 = (const float*)d_in[3];
    const float* b2 = (const float*)d_in[4];
    const float* W3 = (const float*)d_in[5];
    const float* b3 = (const float*)d_in[6];
    float* out = (float*)d_out;

    const int B = in_sizes[0] / (NLEAD * FIN);   // 16384
    const int blocks = B / BT;                   // 2048

    const bool pre = (ws_size >= (size_t)WST_TOTAL * 2u);
    if (pre) {
        __bf16* ws  = (__bf16*)d_ws;
        const __bf16* w1t = ws;
        const __bf16* w2t = ws + W1T_ELEMS;
        const __bf16* w3t = ws + W1T_ELEMS + W2T_ELEMS;
        prep_weights<<<(WST_TOTAL + NTHREADS - 1) / NTHREADS, NTHREADS, 0, stream>>>(
            W1, W2, W3, (__bf16*)d_ws);
        ecg_gcn_fused<true><<<blocks, NTHREADS, 0, stream>>>(
            x, W1, b1, W2, b2, W3, b3, w1t, w2t, w3t, out);
    } else {
        ecg_gcn_fused<false><<<blocks, NTHREADS, 0, stream>>>(
            x, W1, b1, W2, b2, W3, b3, nullptr, nullptr, nullptr, out);
    }
}
